// DTIGraph12PoolLayer_68745246539844
// MI455X (gfx1250) — matrix operand order repaired and checked by reading, not compiled, not run on hardware
//
#include <hip/hip_runtime.h>
#include <hip/hip_bf16.h>
#include <math.h>

typedef __attribute__((ext_vector_type(16))) _Float16 v16h;
typedef __attribute__((ext_vector_type(8)))  float    v8f;

#define DIM 256
#define NNODES 200000
#define NGRAPHS 1024

__device__ __forceinline__ float lrelu(float x) { return x >= 0.f ? x : 0.01f * x; }

__device__ __forceinline__ void atomicMaxFloat(float* addr, float val) {
    // standard split trick: positives via signed max, negatives via unsigned min
    if (val >= 0.f) atomicMax((int*)addr, __float_as_int(val));
    else            atomicMin((unsigned int*)addr, __float_as_uint(val));
}

// Build a 16x32 f16 A-fragment from an f32 row. Per the CDNA5 A layout, lane
// (hi=lane>>4) needs halves[0..7] = K[kc+hi*8 .. +7] and halves[8..15] =
// K[kc+16+hi*8 .. +7] -> two contiguous 8-float runs -> 4x float4 loads.
__device__ __forceinline__ v16h load_a_frag(const float* rp, bool do_leaky) {
    float4 x0 = *(const float4*)(rp);
    float4 x1 = *(const float4*)(rp + 4);
    float4 x2 = *(const float4*)(rp + 16);
    float4 x3 = *(const float4*)(rp + 20);
    float xs[16] = {x0.x,x0.y,x0.z,x0.w, x1.x,x1.y,x1.z,x1.w,
                    x2.x,x2.y,x2.z,x2.w, x3.x,x3.y,x3.z,x3.w};
    v16h a;
#pragma unroll
    for (int i = 0; i < 16; i++) {
        float v = xs[i];
        if (do_leaky) v = lrelu(v);
        a[i] = (_Float16)v;
    }
    return a;
}

// ---------------- kernel 1: per-graph init + gl_dot -------------------------
__global__ void k_init(const float* __restrict__ gf, const float* __restrict__ Wl,
                       float* gl_dot, float* zmax, float* denom, float* g_repr) {
    int g = blockIdx.x, k = threadIdx.x;
    float v = gf[g * DIM + k];
    __shared__ float s[DIM];
    s[k] = lrelu(v) * Wl[k];
    g_repr[g * DIM + k] = 0.f;
    __syncthreads();
    for (int o = 128; o > 0; o >>= 1) { if (k < o) s[k] += s[k + o]; __syncthreads(); }
    if (k == 0) { gl_dot[g] = s[0]; zmax[g] = __uint_as_float(0xFF800000u); denom[g] = 0.f; }
}

// ---------------- kernel 2: weights -> fragment-ordered f16 -----------------
// frag index: ((kt*NT + nt)*32 + lane)*16 + i ; element = B[kt*32 + (lane>>4)*8 + (i&7) + (i>>3)*16][nt*16 + (lane&15)]
__global__ void k_wprep(const float* __restrict__ Wproj, const float* __restrict__ Wih,
                        const float* __restrict__ Whh,
                        _Float16* WprojF, _Float16* WihF, _Float16* WhhF) {
    int t = blockIdx.x * 256 + threadIdx.x;
    if (t < 65536) {                                   // W_proj: B[k][n]=Wproj[k*256+n], NT=16
        int i = t & 15, lane = (t >> 4) & 31, rem = t >> 9;
        int nt = rem & 15, kt = rem >> 4;
        int k = kt * 32 + (lane >> 4) * 8 + (i & 7) + (i >> 3) * 16, n = nt * 16 + (lane & 15);
        WprojF[t] = (_Float16)Wproj[k * DIM + n];
    }
    int t2 = t - 65536;
    if (t2 >= 0 && t2 < 196608) {                      // W_ih^T: B[k][n]=Wih[n*256+k], NT=48
        int i = t2 & 15, lane = (t2 >> 4) & 31, rem = t2 >> 9;
        int nt = rem % 48, kt = rem / 48;
        int k = kt * 32 + (lane >> 4) * 8 + (i & 7) + (i >> 3) * 16, n = nt * 16 + (lane & 15);
        WihF[t2] = (_Float16)Wih[n * DIM + k];
    }
    int t3 = t2 - 196608;
    if (t3 >= 0 && t3 < 196608) {                      // W_hh^T
        int i = t3 & 15, lane = (t3 >> 4) & 31, rem = t3 >> 9;
        int nt = rem % 48, kt = rem / 48;
        int k = kt * 32 + (lane >> 4) * 8 + (i & 7) + (i >> 3) * 16, n = nt * 16 + (lane & 15);
        WhhF[t3] = (_Float16)Whh[n * DIM + k];
    }
}

// ---------------- kernel 3: node logits + segment max -----------------------
__global__ void k_logits(const float* __restrict__ nf, const float* __restrict__ Wl,
                         const float* __restrict__ bl, const int* __restrict__ seg,
                         const float* __restrict__ gl_dot, float* z_e, float* zmax) {
    int lane = threadIdx.x & 31;
    int node = blockIdx.x * 8 + (threadIdx.x >> 5);       // 25000*8 == 200000 exactly
    const float* wp = Wl + DIM + lane * 8;
    float4 w0 = *(const float4*)wp, w1 = *(const float4*)(wp + 4);
    const float* rp = nf + (size_t)node * DIM + lane * 8;
    float4 x0 = *(const float4*)rp, x1 = *(const float4*)(rp + 4);
    float p = x0.x*w0.x + x0.y*w0.y + x0.z*w0.z + x0.w*w0.w
            + x1.x*w1.x + x1.y*w1.y + x1.z*w1.z + x1.w*w1.w;
#pragma unroll
    for (int m = 16; m > 0; m >>= 1) p += __shfl_xor(p, m, 32);
    if (lane == 0) {
        int s = seg[node];
        float zv = lrelu(gl_dot[s] + p + bl[0]);
        z_e[node] = zv;
        atomicMaxFloat(&zmax[s], zv);
    }
}

// ---------------- kernel 4a/4b: softmax -------------------------------------
__global__ void k_softmax_e(const int* __restrict__ seg, float* z_e,
                            const float* __restrict__ zmax, float* denom) {
    int t = blockIdx.x * 256 + threadIdx.x;
    if (t >= NNODES) return;
    int s = seg[t];
    float e = __expf(z_e[t] - zmax[s]);
    z_e[t] = e;
    atomicAdd(&denom[s], e);
}
__global__ void k_softmax_a(const int* __restrict__ seg, const float* __restrict__ z_e,
                            const float* __restrict__ denom, float* a_out) {
    int t = blockIdx.x * 256 + threadIdx.x;
    if (t >= NNODES) return;
    a_out[t] = z_e[t] / denom[seg[t]];
}

// ---------------- kernel 5: WMMA projection + attention pooling -------------
// block = 256 thr = 8 waves; chunk = 64 rows x 256 cols; K = 256 (8 steps of 32).
// Wave w: M-tile = w&3 (16 rows), col half = w>>2 (8 N-tiles). W_proj fragments
// staged in 128KB LDS once per block; grid-stride over 3125 chunks amortizes it.
// NOTE: kt loop kept rolled (#pragma unroll 1) so the 8x8=64 B-fragment LDS
// loads cannot be hoisted out of the chunk loop by LICM -> no scratch spills;
// B fragments stream from LDS (ds_load_b128) right before each v_wmma.
__global__ __launch_bounds__(256) void k_proj_pool(
    const float* __restrict__ nf, const float* __restrict__ b_proj,
    const int* __restrict__ seg, const float* __restrict__ a_out,
    const _Float16* __restrict__ WprojF, float* g_repr) {
    extern __shared__ _Float16 sW[];                       // 65536 halves = 128 KB
    {
        const uint4* src = (const uint4*)WprojF;
        uint4* dst = (uint4*)sW;
        for (int i = threadIdx.x; i < 8192; i += 256) dst[i] = src[i];
    }
    __syncthreads();

    int w = threadIdx.x >> 5, lane = threadIdx.x & 31;
    int lo = lane & 15, hi = lane >> 4;
    int mt = w & 3, nh = w >> 2;
    v8f zero = {};

    for (int chunk = blockIdx.x; chunk < NNODES / 64; chunk += gridDim.x) {
        int rowbase = chunk * 64 + mt * 16;
        v8f acc[8];
#pragma unroll
        for (int j = 0; j < 8; j++) acc[j] = zero;

#pragma unroll 1
        for (int kt = 0; kt < 8; kt++) {
            const float* rp = nf + (size_t)(rowbase + lo) * DIM + kt * 32 + hi * 8;
            v16h a = load_a_frag(rp, false);
#pragma unroll
            for (int j = 0; j < 8; j++) {
                int nt = nh * 8 + j;
                v16h b = *(const v16h*)(sW + ((kt * 16 + nt) * 512 + lane * 16));
                acc[j] = __builtin_amdgcn_wmma_f32_16x16x32_f16(
                    false, a, false, b, (short)0, acc[j], false, false);
            }
        }
        // epilogue: C elem (M = r + 8*hi, N = 16*nt + lo); rows per lane fixed.
        int r0 = rowbase + hi * 8;
        float av[8]; int sg[8];
#pragma unroll
        for (int r = 0; r < 8; r++) { av[r] = a_out[r0 + r]; sg[r] = seg[r0 + r]; }
        bool same = (sg[0] == sg[7]);                     // sorted segments: usually true
#pragma unroll
        for (int j = 0; j < 8; j++) {
            int n = (nh * 8 + j) * 16 + lo;
            float bp = b_proj[n];
            if (same) {
                float s = 0.f;
#pragma unroll
                for (int r = 0; r < 8; r++) s += av[r] * lrelu(acc[j][r] + bp);
                atomicAdd(&g_repr[sg[0] * DIM + n], s);
            } else {
#pragma unroll
                for (int r = 0; r < 8; r++)
                    atomicAdd(&g_repr[sg[r] * DIM + n], av[r] * lrelu(acc[j][r] + bp));
            }
        }
    }
}

// ---------------- kernel 6: GRU input/hidden GEMMs (WMMA) -------------------
// grid = (16 Mchunks, 3 col-groups of 256, 2 {gi,gh}); block 64 rows x 256 cols.
__global__ __launch_bounds__(256) void k_gru_gemm(
    const float* __restrict__ g_repr, const float* __restrict__ g_feats,
    const _Float16* __restrict__ WihF, const _Float16* __restrict__ WhhF,
    const float* __restrict__ b_ih, const float* __restrict__ b_hh,
    float* gi, float* gh) {
    extern __shared__ _Float16 sW[];                       // 65536 halves
    int which = blockIdx.z;
    const _Float16* Wf = which ? WhhF : WihF;
    const float* bias  = which ? b_hh : b_ih;
    const float* Asrc  = which ? g_feats : g_repr;         // context gets leaky below
    float* out = which ? gh : gi;
    int ntBase = blockIdx.y * 16;
    for (int d8 = threadIdx.x; d8 < 8192; d8 += 256) {     // restripe NT=48 -> NT=16 slab
        int i8 = d8 & 1, lane = (d8 >> 1) & 31, rem = d8 >> 6;
        int nt = rem & 15, kt = rem >> 4;
        ((uint4*)sW)[d8] = ((const uint4*)Wf)[((kt * 48 + ntBase + nt) * 32 + lane) * 2 + i8];
    }
    __syncthreads();

    int w = threadIdx.x >> 5, lane = threadIdx.x & 31;
    int lo = lane & 15, hi = lane >> 4;
    int mt = w & 3, nh = w >> 2;
    int rowbase = blockIdx.x * 64 + mt * 16;
    v8f zero = {};
    v8f acc[8];
#pragma unroll
    for (int j = 0; j < 8; j++) acc[j] = zero;

    // rolled kt loop: bounds register pressure, keeps B-frag ds_loads in-loop
#pragma unroll 1
    for (int kt = 0; kt < 8; kt++) {
        const float* rp = Asrc + (size_t)(rowbase + lo) * DIM + kt * 32 + hi * 8;
        v16h a = load_a_frag(rp, which == 0);              // context = leaky(g_repr)
#pragma unroll
        for (int j = 0; j < 8; j++) {
            int nt = nh * 8 + j;
            v16h b = *(const v16h*)(sW + ((kt * 16 + nt) * 512 + lane * 16));
            acc[j] = __builtin_amdgcn_wmma_f32_16x16x32_f16(
                false, a, false, b, (short)0, acc[j], false, false);
        }
    }
    int r0 = rowbase + hi * 8;
#pragma unroll
    for (int j = 0; j < 8; j++) {
        int n = (ntBase + nh * 8 + j) * 16 + lo;
        float bp = bias[n];
#pragma unroll
        for (int r = 0; r < 8; r++) out[(size_t)(r0 + r) * 768 + n] = acc[j][r] + bp;
    }
}

// ---------------- kernel 7: GRU gates + final leaky -------------------------
__global__ void k_gru_gates(const float* __restrict__ gi, const float* __restrict__ gh,
                            const float* __restrict__ g_feats, float* outp) {
    int t = blockIdx.x * 256 + threadIdx.x;                // 1024*256
    int m = t >> 8, j = t & 255;
    const float* gim = gi + (size_t)m * 768;
    const float* ghm = gh + (size_t)m * 768;
    float r  = 1.f / (1.f + __expf(-(gim[j]       + ghm[j])));
    float zz = 1.f / (1.f + __expf(-(gim[256 + j] + ghm[256 + j])));
    float nn = tanhf(gim[512 + j] + r * ghm[512 + j]);
    float o = (1.f - zz) * nn + zz * g_feats[t];
    outp[t] = lrelu(o);
}

extern "C" void kernel_launch(void* const* d_in, const int* in_sizes, int n_in,
                              void* d_out, int out_size, void* d_ws, size_t ws_size,
                              hipStream_t stream) {
    const float* nf      = (const float*)d_in[0];
    const float* gf      = (const float*)d_in[1];
    const int*   seg     = (const int*)  d_in[2];
    const float* W_logit = (const float*)d_in[3];
    const float* b_logit = (const float*)d_in[4];
    const float* W_proj  = (const float*)d_in[5];
    const float* b_proj  = (const float*)d_in[6];
    const float* W_ih    = (const float*)d_in[7];
    const float* W_hh    = (const float*)d_in[8];
    const float* b_ih    = (const float*)d_in[9];
    const float* b_hh    = (const float*)d_in[10];

    float* out_gru = (float*)d_out;               // (1024,256)
    float* a_out   = (float*)d_out + NGRAPHS*DIM; // (200000,1)

    // workspace carve-up (~8.7 MB)
    float* z_e    = (float*)d_ws;                 // 200000
    float* zmax   = z_e + NNODES;                 // 1024
    float* denom  = zmax + NGRAPHS;               // 1024
    float* gl_dot = denom + NGRAPHS;              // 1024
    float* g_repr = gl_dot + NGRAPHS;             // 262144
    float* gi     = g_repr + NGRAPHS*DIM;         // 786432
    float* gh     = gi + NGRAPHS*768;             // 786432
    _Float16* WprojF = (_Float16*)(gh + NGRAPHS*768); // 65536 halves (32B aligned)
    _Float16* WihF   = WprojF + 65536;            // 196608 halves
    _Float16* WhhF   = WihF + 196608;             // 196608 halves

    k_init<<<NGRAPHS, 256, 0, stream>>>(gf, W_logit, gl_dot, zmax, denom, g_repr);
    k_wprep<<<(65536 + 2*196608) / 256, 256, 0, stream>>>(W_proj, W_ih, W_hh,
                                                          WprojF, WihF, WhhF);
    k_logits<<<NNODES / 8, 256, 0, stream>>>(nf, W_logit, b_logit, seg, gl_dot, z_e, zmax);
    k_softmax_e<<<(NNODES + 255) / 256, 256, 0, stream>>>(seg, z_e, zmax, denom);
    k_softmax_a<<<(NNODES + 255) / 256, 256, 0, stream>>>(seg, z_e, denom, a_out);
    k_proj_pool<<<1024, 256, 131072, stream>>>(nf, b_proj, seg, a_out, WprojF, g_repr);
    k_gru_gemm<<<dim3(16, 3, 2), 256, 131072, stream>>>(g_repr, gf, WihF, WhhF,
                                                        b_ih, b_hh, gi, gh);
    k_gru_gates<<<NGRAPHS, 256, 0, stream>>>(gi, gh, gf, out_gru);
}
